// desperate_transformer_14826227106168
// MI455X (gfx1250) — matrix operand order repaired and checked by reading, not compiled, not run on hardware
//
#include <hip/hip_runtime.h>
#include <math.h>

// ---------------------------------------------------------------------------
// CDNA5 (gfx1250) implementation of the reference transformer sampler.
// GEMMs run on matrix cores via v_wmma_f32_16x16x32_f16 (f16 in, f32 acc).
// Weights (~128MB f32, ~67MB f16 copy) fit in the 192MB L2, so the 35-step
// decode loop is L2-resident. Weights are converted once per launch into
// zero-padded f16 (N->mult of 64, K->mult of 32) so the WMMA inner loop has
// NO bounds checks and only wide 16B vector loads.
//
// Input flattening assumption (recursive dict-insertion order):
//   0:x  1:x2  2..23:f1  24..45:f2  46..67:f3  68..89:s1  90..111:s2
//   112..133:s3  134:emb  135:lmW  136:lmb
// Each block: key(W1,b1,W2,b2,W3,b3) val(...) query(...) mha(Wi,bi,Wo,bo)
// Workspace requirement: ~220 MB.
// ---------------------------------------------------------------------------

#define B_   128
#define T_   35
#define IND_ 2048
#define H_   512
#define NH_  8
#define FC_  600
#define V_   10000
#define DIN_ (H_ + T_)   // 547
#define EFP_ 576         // DIN_ padded to 32
#define FCP_ 608         // FC_ padded to 32

typedef __attribute__((ext_vector_type(16))) _Float16 v16h;
typedef __attribute__((ext_vector_type(8)))  _Float16 v8h;
typedef __attribute__((ext_vector_type(8)))  float    v8f;
typedef __attribute__((ext_vector_type(4)))  float    v4f;

// ------------------------------ Threefry-2x32-20 ---------------------------
__host__ __device__ inline void tf2x32(unsigned k0, unsigned k1,
                                       unsigned x0, unsigned x1,
                                       unsigned &o0, unsigned &o1) {
  unsigned ks2 = k0 ^ k1 ^ 0x1BD11BDAu;
  x0 += k0; x1 += k1;
#define TF_RND(r) { x0 += x1; x1 = (x1 << (r)) | (x1 >> (32 - (r))); x1 ^= x0; }
  TF_RND(13) TF_RND(15) TF_RND(26) TF_RND(6)  x0 += k1;  x1 += ks2 + 1u;
  TF_RND(17) TF_RND(29) TF_RND(16) TF_RND(24) x0 += ks2; x1 += k0 + 2u;
  TF_RND(13) TF_RND(15) TF_RND(26) TF_RND(6)  x0 += k0;  x1 += k1 + 3u;
  TF_RND(17) TF_RND(29) TF_RND(16) TF_RND(24) x0 += k1;  x1 += ks2 + 4u;
  TF_RND(13) TF_RND(15) TF_RND(26) TF_RND(6)  x0 += ks2; x1 += k0 + 5u;
#undef TF_RND
  o0 = x0; o1 = x1;
}

// --------------------- weight f32 -> padded f16 convert --------------------
__global__ __launch_bounds__(256)
void cvt_pad_f16(const float* __restrict__ W, _Float16* __restrict__ Wh,
                 int N, int K, int Np, int Kp)
{
  long idx = (long)blockIdx.x * 256 + threadIdx.x;
  long total = (long)Np * Kp;
  if (idx >= total) return;
  int k = (int)(idx % Kp);
  long n = idx / Kp;
  Wh[idx] = (n < N && k < K) ? (_Float16)W[n * K + k] : (_Float16)0.0f;
}

__global__ void zero_buf(float* __restrict__ p, long n) {
  long i = (long)blockIdx.x * 256 + threadIdx.x;
  if (i < n) p[i] = 0.0f;
}

// ------------------------------ WMMA GEMM ----------------------------------
// C[M,N](ldc) = act(A[M,>=Kp](lda) @ Wh[Npad,Kp]^T + bias[N]); act 1 = relu.
// Wh is zero-padded f16: inner loop is guard-free (A pad cols * 0 = 0).
// One wave computes 16(M) x 64(N); workgroup = 4 waves along M.
// Fragment layouts per CDNA5 ISA 7.12.2 (wave32):
//  A 16x32 f16 : lane L row M=L&15; K = k0 + h8 + j (j<8), k0+16+h8+j (j>=8),
//                h8 = (L>=16)?8:0  -> two contiguous 8-float runs.
//  B 32x16 f16 : lane L col N=L&15; K same as A (h8 + j, 16 + h8 + j)
//                -> two contiguous 8-half runs of one weight row.
//  C 16x16 f32 : lane L col N=L&15; VGPR r -> row M = r + ((L>=16)?8:0).
// Alignment: all A bases/strides and Wh strides are multiples of 16 bytes.
__global__ __launch_bounds__(128)
void gemm_f16w(const float* __restrict__ A, long lda,
               const _Float16* __restrict__ Wh, int Kp,
               const float* __restrict__ bias,
               float* __restrict__ C, int ldc,
               int M, int N, int act)
{
  const int lane   = threadIdx.x & 31;
  const int wave   = threadIdx.x >> 5;
  const int m0     = blockIdx.x * 64 + wave * 16;
  const int n0     = blockIdx.y * 64;
  const int half16 = lane >> 4;          // 0 / 1
  const int ncol   = lane & 15;
  if (m0 >= M) return;

  const float* arow = A + (long)(m0 + (lane & 15)) * lda + (half16 << 3);
  const _Float16* wr0 = Wh + (long)(n0 + ncol) * Kp + (half16 << 3);
  const _Float16* wr1 = wr0 + (long)16 * Kp;
  const _Float16* wr2 = wr0 + (long)32 * Kp;
  const _Float16* wr3 = wr0 + (long)48 * Kp;

  v8f acc[4] = {};
  for (int k0 = 0; k0 < Kp; k0 += 32) {
    v4f a0 = *(const v4f*)(arow + k0);
    v4f a1 = *(const v4f*)(arow + k0 + 4);
    v4f a2 = *(const v4f*)(arow + k0 + 16);
    v4f a3 = *(const v4f*)(arow + k0 + 20);
    v8h alo = __builtin_convertvector(
        __builtin_shufflevector(a0, a1, 0,1,2,3,4,5,6,7), v8h);
    v8h ahi = __builtin_convertvector(
        __builtin_shufflevector(a2, a3, 0,1,2,3,4,5,6,7), v8h);
    v16h af = __builtin_shufflevector(alo, ahi,
        0,1,2,3,4,5,6,7,8,9,10,11,12,13,14,15);

    v8h b0l = *(const v8h*)(wr0 + k0);
    v8h b0h = *(const v8h*)(wr0 + k0 + 16);
    v16h bf0 = __builtin_shufflevector(b0l, b0h,
        0,1,2,3,4,5,6,7,8,9,10,11,12,13,14,15);
    acc[0] = __builtin_amdgcn_wmma_f32_16x16x32_f16(
        false, af, false, bf0, (short)0, acc[0], false, false);

    v8h b1l = *(const v8h*)(wr1 + k0);
    v8h b1h = *(const v8h*)(wr1 + k0 + 16);
    v16h bf1 = __builtin_shufflevector(b1l, b1h,
        0,1,2,3,4,5,6,7,8,9,10,11,12,13,14,15);
    acc[1] = __builtin_amdgcn_wmma_f32_16x16x32_f16(
        false, af, false, bf1, (short)0, acc[1], false, false);

    v8h b2l = *(const v8h*)(wr2 + k0);
    v8h b2h = *(const v8h*)(wr2 + k0 + 16);
    v16h bf2 = __builtin_shufflevector(b2l, b2h,
        0,1,2,3,4,5,6,7,8,9,10,11,12,13,14,15);
    acc[2] = __builtin_amdgcn_wmma_f32_16x16x32_f16(
        false, af, false, bf2, (short)0, acc[2], false, false);

    v8h b3l = *(const v8h*)(wr3 + k0);
    v8h b3h = *(const v8h*)(wr3 + k0 + 16);
    v16h bf3 = __builtin_shufflevector(b3l, b3h,
        0,1,2,3,4,5,6,7,8,9,10,11,12,13,14,15);
    acc[3] = __builtin_amdgcn_wmma_f32_16x16x32_f16(
        false, af, false, bf3, (short)0, acc[3], false, false);
  }

  const int mbase = m0 + (half16 << 3);
#pragma unroll
  for (int nt = 0; nt < 4; ++nt) {
    int n = n0 + nt * 16 + ncol;
    if (n >= N) continue;
    float bv = bias[n];
#pragma unroll
    for (int r = 0; r < 8; ++r) {
      float v = acc[nt][r] + bv;
      if (act) v = fmaxf(v, 0.0f);
      C[(long)(mbase + r) * ldc + n] = v;
    }
  }
}

// ------------------------------ row softmax --------------------------------
__global__ __launch_bounds__(256)
void row_softmax(float* __restrict__ X, int N) {
  __shared__ float sm[8];
  float* x = X + (long)blockIdx.x * N;
  const int tid = threadIdx.x;
  float mx = -__builtin_inff();
  for (int j = tid; j < N; j += 256) mx = fmaxf(mx, x[j]);
  for (int off = 16; off; off >>= 1) mx = fmaxf(mx, __shfl_xor(mx, off, 32));
  if ((tid & 31) == 0) sm[tid >> 5] = mx;
  __syncthreads();
  if (tid == 0) { float m = sm[0]; for (int j = 1; j < 8; ++j) m = fmaxf(m, sm[j]); sm[0] = m; }
  __syncthreads();
  mx = sm[0];
  __syncthreads();
  float s = 0.0f;
  for (int j = tid; j < N; j += 256) s += expf(x[j] - mx);
  for (int off = 16; off; off >>= 1) s += __shfl_xor(s, off, 32);
  if ((tid & 31) == 0) sm[tid >> 5] = s;
  __syncthreads();
  if (tid == 0) { float t = 0.0f; for (int j = 1; j < 8; ++j) t += sm[j]; sm[0] += t; }
  __syncthreads();
  float inv = 1.0f / sm[0];
  for (int j = tid; j < N; j += 256) x[j] = expf(x[j] - mx) * inv;
}

// ------------------- fused attention (online softmax) ----------------------
// grid = (Tq, NH, Bz), block = 32 (one wave). Rows: (b*T + t)*H + h*64.
// kmax = number of valid key positions (mask arange(T) <= i -> kmax = i+1).
__global__ __launch_bounds__(32)
void attn_fused(const float* __restrict__ QP, const float* __restrict__ KP,
                const float* __restrict__ VP, float* __restrict__ AO,
                int T, int kmax)
{
  const int q = blockIdx.x, h = blockIdx.y, b = blockIdx.z;
  const int lane = threadIdx.x;
  const long base = ((long)b * T) * H_ + h * 64;
  const float scale = 0.125f;   // 1/sqrt(64)
  const long qr = base + (long)q * H_;
  float qa = QP[qr + lane] * scale;
  float qb = QP[qr + lane + 32] * scale;
  float m = -__builtin_inff(), l = 0.0f, a0 = 0.0f, a1 = 0.0f;
  for (int k = 0; k < kmax; ++k) {
    const long kr = base + (long)k * H_;
    float s = qa * KP[kr + lane] + qb * KP[kr + lane + 32];
    for (int off = 16; off; off >>= 1) s += __shfl_xor(s, off, 32);
    float mn = fmaxf(m, s);
    float c = expf(m - mn);
    float p = expf(s - mn);
    l  = l  * c + p;
    a0 = a0 * c + p * VP[kr + lane];
    a1 = a1 * c + p * VP[kr + lane + 32];
    m = mn;
  }
  float inv = 1.0f / l;
  AO[qr + lane]      = a0 * inv;
  AO[qr + lane + 32] = a1 * inv;
}

// --------------------------- misc small kernels ----------------------------
__global__ void init_seq(int* __restrict__ seq) {
  int i = blockIdx.x * 256 + threadIdx.x;
  if (i < B_ * T_) seq[i] = 1;
}

// EI/FI have padded row stride EFP_; pad cols pre-zeroed once per launch.
__global__ __launch_bounds__(256)
void build_inputs(const float* __restrict__ emb, const int* __restrict__ seq,
                  const float* __restrict__ feats,
                  float* __restrict__ EI, float* __restrict__ FI, int step)
{
  long idx = (long)blockIdx.x * 256 + threadIdx.x;
  const long total = (long)B_ * T_ * DIN_;
  if (idx >= total) return;
  int c = (int)(idx % DIN_);
  long bt = idx / DIN_;
  long o = bt * EFP_ + c;
  if (c < H_) {
    int tok = seq[bt];
    EI[o] = emb[(long)tok * H_ + c];
    FI[o] = feats[(bt / T_) * H_ + c];
  } else {
    float pv = (c - H_ == step) ? 1.0f : 0.0f;
    EI[o] = pv;
    FI[o] = pv;
  }
}

__global__ void add2(float* __restrict__ o, const float* __restrict__ a,
                     const float* __restrict__ b, long n) {
  long i = (long)blockIdx.x * 256 + threadIdx.x;
  if (i < n) o[i] = a[i] + b[i];
}
__global__ void add3(float* __restrict__ o, const float* __restrict__ a,
                     const float* __restrict__ b, const float* __restrict__ c, long n) {
  long i = (long)blockIdx.x * 256 + threadIdx.x;
  if (i < n) o[i] = a[i] + b[i] + c[i];
}

// ---------------- fused logits softmax + threefry-Gumbel sample ------------
__global__ __launch_bounds__(256)
void softmax_sample(const float* __restrict__ LG, float* __restrict__ out,
                    int* __restrict__ seq, int step, unsigned k0, unsigned k1)
{
  __shared__ float sm[8];
  __shared__ float sbv[8];
  __shared__ int   sbi[8];
  const int b = blockIdx.x, tid = threadIdx.x;
  const float* l = LG + (long)b * V_;
  float mx = -__builtin_inff();
  for (int v = tid; v < V_; v += 256) mx = fmaxf(mx, l[v]);
  for (int off = 16; off; off >>= 1) mx = fmaxf(mx, __shfl_xor(mx, off, 32));
  if ((tid & 31) == 0) sm[tid >> 5] = mx;
  __syncthreads();
  if (tid == 0) { float m = sm[0]; for (int j = 1; j < 8; ++j) m = fmaxf(m, sm[j]); sm[0] = m; }
  __syncthreads();
  mx = sm[0];
  __syncthreads();
  float s = 0.0f;
  for (int v = tid; v < V_; v += 256) s += expf(l[v] - mx);
  for (int off = 16; off; off >>= 1) s += __shfl_xor(s, off, 32);
  if ((tid & 31) == 0) sm[tid >> 5] = s;
  __syncthreads();
  if (tid == 0) { float t = 0.0f; for (int j = 1; j < 8; ++j) t += sm[j]; sm[0] += t; }
  __syncthreads();
  const float inv = 1.0f / sm[0];
  float* orow = out + ((long)b * T_ + step) * V_;
  for (int v = tid; v < V_; v += 256) orow[v] = expf(l[v] - mx) * inv;
  // Gumbel-argmax sampling (jax.random.categorical semantics)
  const long half = (long)B_ * V_ / 2;
  float bv = -__builtin_inff(); int bi = 0;
  for (int v = tid; v < V_; v += 256) {
    long e = (long)b * V_ + v;
    unsigned r0, r1, bits;
    if (e < half) { tf2x32(k0, k1, (unsigned)e, (unsigned)(e + half), r0, r1); bits = r0; }
    else          { tf2x32(k0, k1, (unsigned)(e - half), (unsigned)e, r0, r1); bits = r1; }
    float u = __uint_as_float((bits >> 9) | 0x3f800000u) - 1.0f;
    u = fmaxf(u, 1.1754944e-38f);
    float g = -logf(-logf(u));
    float val = l[v] + g;
    if (val > bv || (val == bv && v < bi)) { bv = val; bi = v; }
  }
  for (int off = 16; off; off >>= 1) {
    float ov = __shfl_xor(bv, off, 32);
    int   oi = __shfl_xor(bi, off, 32);
    if (ov > bv || (ov == bv && oi < bi)) { bv = ov; bi = oi; }
  }
  if ((tid & 31) == 0) { sbv[tid >> 5] = bv; sbi[tid >> 5] = bi; }
  __syncthreads();
  if (tid == 0) {
    float best = sbv[0]; int bidx = sbi[0];
    for (int j = 1; j < 8; ++j)
      if (sbv[j] > best || (sbv[j] == best && sbi[j] < bidx)) { best = sbv[j]; bidx = sbi[j]; }
    if (step + 1 < T_) seq[b * T_ + step + 1] = bidx;
  }
}

// ------------------------------ host-side ----------------------------------
struct WH { const _Float16* w; int kp; };
struct BlockH {
  WH kW1,kW2,kW3, vW1,vW2,vW3, qW1,qW2,qW3, Wi, Wo;
  const float *kb1,*kb2,*kb3, *vb1,*vb2,*vb3, *qb1,*qb2,*qb3, *bi,*bo;
};

static WH prep(hipStream_t st, _Float16*& cur, const float* W, int N, int K) {
  int Kp = (K + 31) & ~31;
  int Np = (N + 63) & ~63;
  _Float16* dst = cur;
  long total = (long)Np * Kp;
  cur += total;
  cvt_pad_f16<<<(int)((total + 255) / 256), 256, 0, st>>>(W, dst, N, K, Np, Kp);
  WH r = {dst, Kp};
  return r;
}

// d: f32 leaf pointers of one block (key W1,b1,W2,b2,W3,b3 | val | query | mha)
static BlockH prepBlock(hipStream_t st, _Float16*& cur, void* const* d, int base, int in_d) {
  auto F = [&](int k){ return (const float*)d[base + k]; };
  BlockH P;
  P.kW1 = prep(st, cur, F(0),  FC_, in_d); P.kb1 = F(1);
  P.kW2 = prep(st, cur, F(2),  FC_, FC_);  P.kb2 = F(3);
  P.kW3 = prep(st, cur, F(4),  H_,  FC_);  P.kb3 = F(5);
  P.vW1 = prep(st, cur, F(6),  FC_, in_d); P.vb1 = F(7);
  P.vW2 = prep(st, cur, F(8),  FC_, FC_);  P.vb2 = F(9);
  P.vW3 = prep(st, cur, F(10), H_,  FC_);  P.vb3 = F(11);
  P.qW1 = prep(st, cur, F(12), FC_, in_d); P.qb1 = F(13);
  P.qW2 = prep(st, cur, F(14), FC_, FC_);  P.qb2 = F(15);
  P.qW3 = prep(st, cur, F(16), H_,  FC_);  P.qb3 = F(17);
  P.Wi  = prep(st, cur, F(18), 3 * H_, H_); P.bi = F(19);
  P.Wo  = prep(st, cur, F(20), H_,  H_);   P.bo = F(21);
  return P;
}

struct WsBufs { float *H1,*H2,*QM,*KM,*VM,*QP,*KP,*VP,*AO; };

static inline void gemmL(hipStream_t st, const float* A, long lda, WH W,
                         const float* b, float* C, int ldc, int M, int N, int act) {
  dim3 grid(M / 64, (N + 63) / 64);
  gemm_f16w<<<grid, 128, 0, st>>>(A, lda, W.w, W.kp, b, C, ldc, M, N, act);
}

// One attention block: out = MHA(softmaxMLP(Q), reluMLP(K), reluMLP(V)).
static void run_block(hipStream_t st, const BlockH& P,
                      const float* Kin, const float* Vin, const float* Qin, long ld_in,
                      int M, int Tattn, int Bz, int kmax,
                      float* outBuf, const WsBufs& w)
{
  // query MLP (softmax output)
  gemmL(st, Qin,  ld_in, P.qW1, P.qb1, w.H1, FCP_, M, FC_, 1);
  gemmL(st, w.H1, FCP_,  P.qW2, P.qb2, w.H2, FCP_, M, FC_, 1);
  gemmL(st, w.H2, FCP_,  P.qW3, P.qb3, w.QM, H_,   M, H_,  0);
  row_softmax<<<M, 256, 0, st>>>(w.QM, H_);
  // key MLP (relu output)
  gemmL(st, Kin,  ld_in, P.kW1, P.kb1, w.H1, FCP_, M, FC_, 1);
  gemmL(st, w.H1, FCP_,  P.kW2, P.kb2, w.H2, FCP_, M, FC_, 1);
  gemmL(st, w.H2, FCP_,  P.kW3, P.kb3, w.KM, H_,   M, H_,  1);
  // value MLP (relu output)
  gemmL(st, Vin,  ld_in, P.vW1, P.vb1, w.H1, FCP_, M, FC_, 1);
  gemmL(st, w.H1, FCP_,  P.vW2, P.vb2, w.H2, FCP_, M, FC_, 1);
  gemmL(st, w.H2, FCP_,  P.vW3, P.vb3, w.VM, H_,   M, H_,  1);
  // QKV projections: rows [0,H), [H,2H), [2H,3H) of padded Wi (stride kp)
  WH WiQ = {P.Wi.w,                       P.Wi.kp};
  WH WiK = {P.Wi.w + (long)H_ * P.Wi.kp,  P.Wi.kp};
  WH WiV = {P.Wi.w + 2L * H_ * P.Wi.kp,   P.Wi.kp};
  gemmL(st, w.QM, H_, WiQ, P.bi,          w.QP, H_, M, H_, 0);
  gemmL(st, w.KM, H_, WiK, P.bi + H_,     w.KP, H_, M, H_, 0);
  gemmL(st, w.VM, H_, WiV, P.bi + 2 * H_, w.VP, H_, M, H_, 0);
  // attention + output projection
  attn_fused<<<dim3(Tattn, NH_, Bz), 32, 0, st>>>(w.QP, w.KP, w.VP, w.AO, Tattn, kmax);
  gemmL(st, w.AO, H_, P.Wo, P.bo, outBuf, H_, M, H_, 0);
}

extern "C" void kernel_launch(void* const* d_in, const int* in_sizes, int n_in,
                              void* d_out, int out_size, void* d_ws, size_t ws_size,
                              hipStream_t stream)
{
  (void)in_sizes; (void)n_in; (void)out_size; (void)ws_size;
  const float* x   = (const float*)d_in[0];
  const float* emb = (const float*)d_in[134];
  const float* lmW = (const float*)d_in[135];
  const float* lmb = (const float*)d_in[136];
  float* out = (float*)d_out;

  // ---- carve f32 workspace ----
  float* wp = (float*)d_ws;
  auto take = [&](long n) { float* p = wp; wp += n; return p; };
  const long M2 = (long)B_ * T_;                 // 4480
  float* EI  = take(M2 * EFP_);                  // padded DIN rows
  float* FI  = take(M2 * EFP_);
  float* H1  = take(M2 * FCP_);                  // padded FC rows
  float* H2  = take(M2 * FCP_);
  float* QM  = take(M2 * H_);
  float* KM  = take(M2 * H_);
  float* VM  = take(M2 * H_);
  float* QP  = take(M2 * H_);
  float* KP  = take(M2 * H_);
  float* VP  = take(M2 * H_);
  float* AO  = take(M2 * H_);
  float* S1b = take(M2 * H_);
  float* S2b = take(M2 * H_);
  float* S3b = take(M2 * H_);
  float* TMP = take(M2 * H_);
  float* O1  = take((long)B_ * H_);
  float* O2  = take((long)B_ * H_);
  float* FE  = take((long)B_ * H_);
  float* LG  = take((long)B_ * V_);
  int*   SEQ = (int*)wp;
  wp += (B_ * T_ + 4);
  // ---- f16 weight cache region ----
  _Float16* hcur = (_Float16*)wp;

  // zero EI/FI/H1/H2 (contiguous) so pad columns are clean even pre-poison
  {
    long nz = 2 * M2 * EFP_ + 2 * M2 * FCP_;
    zero_buf<<<(int)((nz + 255) / 256), 256, 0, stream>>>(EI, nz);
  }
  init_seq<<<(B_ * T_ + 255) / 256, 256, 0, stream>>>(SEQ);

  // convert all weights to padded f16 (once per launch; reused 35x)
  BlockH f1 = prepBlock(stream, hcur, d_in, 2,   IND_);
  BlockH f2 = prepBlock(stream, hcur, d_in, 24,  H_);
  BlockH f3 = prepBlock(stream, hcur, d_in, 46,  H_);
  BlockH s1 = prepBlock(stream, hcur, d_in, 68,  DIN_);
  BlockH s2 = prepBlock(stream, hcur, d_in, 90,  H_);
  BlockH s3 = prepBlock(stream, hcur, d_in, 112, H_);
  WH     lm = prep(stream, hcur, lmW, V_, H_);

  WsBufs wb = {H1, H2, QM, KM, VM, QP, KP, VP, AO};

  // ---- Phase 1: feature extractor (batch 1, sequence length B_) ----
  const long nSmall = (long)B_ * H_;
  run_block(stream, f1, x,  x,  x,  IND_, B_, B_, 1, B_, O1,  wb);
  run_block(stream, f2, O1, O1, O1, H_,   B_, B_, 1, B_, TMP, wb);
  add2<<<(int)((nSmall + 255) / 256), 256, 0, stream>>>(O2, TMP, O1, nSmall);
  run_block(stream, f3, O2, O2, O2, H_,   B_, B_, 1, B_, TMP, wb);
  add3<<<(int)((nSmall + 255) / 256), 256, 0, stream>>>(FE, TMP, O2, O1, nSmall);

  // ---- Phase 2: autoregressive decode, T_ steps ----
  const long nBig   = M2 * H_;
  const long nBuild = M2 * DIN_;
  for (int i = 0; i < T_; ++i) {
    build_inputs<<<(int)((nBuild + 255) / 256), 256, 0, stream>>>(emb, SEQ, FE, EI, FI, i);
    run_block(stream, s1, EI,  FI,  EI,  EFP_, (int)M2, T_, B_, i + 1, S1b, wb);
    run_block(stream, s2, S1b, S1b, S1b, H_,   (int)M2, T_, B_, i + 1, TMP, wb);
    add2<<<(int)((nBig + 255) / 256), 256, 0, stream>>>(S2b, TMP, S1b, nBig);
    run_block(stream, s3, S2b, S2b, S2b, H_,   (int)M2, T_, B_, i + 1, TMP, wb);
    add3<<<(int)((nBig + 255) / 256), 256, 0, stream>>>(S3b, TMP, S2b, S1b, nBig);
    // logits from position i only: rows stride T_*H_
    gemmL(stream, S3b + (long)i * H_, (long)T_ * H_, lm, lmb, LG, V_, B_, V_, 0);
    // fold_in(key(42), i) computed on host (deterministic, capture-safe)
    unsigned fk0, fk1;
    tf2x32(0u, 42u, 0u, (unsigned)i, fk0, fk1);
    softmax_sample<<<B_, 256, 0, stream>>>(LG, out, SEQ, i, fk0, fk1);
  }
}
